// MusicGeneration_64304250355968
// MI455X (gfx1250) — matrix operand order repaired and checked by reading, not compiled, not run on hardware
//
#include <hip/hip_runtime.h>
#include <hip/hip_bf16.h>

// ---------------------------------------------------------------------------
// Biaxial music-gen LSTM on MI455X (gfx1250).
// Stage 1: time LSTM (H=128, T=128, batch 2496)  -> f16 WMMA, weight-stationary
// Stage 2: note LSTM (NH=2, 78 steps, batch 4096) -> f16 WMMA, 1 wave / 16 seqs
// Workspace: th (time-LSTM hidden states) stored f16: 2496*128*128*2 = ~78 MB.
//
// v2: software-pipeline the A fragments — issue all 14 ds_load_b128 for the
// step's 7 A fragments before the 28-WMMA chain, instead of interleaving
// load->wait->wmma per fragment (removes 6 full s_wait_dscnt stalls from the
// recurrent critical path).
// ---------------------------------------------------------------------------

typedef __attribute__((ext_vector_type(16))) _Float16 v16h;
typedef __attribute__((ext_vector_type(8)))  _Float16 v8h;
typedef __attribute__((ext_vector_type(8)))  float    v8f;

#define BATCH 32
#define NOTES 78
#define TT    128
#define IN_F  80
#define HID   128
#define NHID  2
#define NSEQ1 (BATCH * NOTES)   // 2496
#define G1    (4 * HID)         // 512 gate columns
#define KX    96                // IN_F padded to multiple of 32
#define NSEQ2 (BATCH * TT)      // 4096

__device__ __forceinline__ float sigf(float v) { return 1.0f / (1.0f + __expf(-v)); }

union HF16x16 { v16h v; v8h h[2]; };

// 16-bit A fragment (16x32, M x K), row-major source with row = lane&15:
// halves 0..7  <- K = kb + 16*hi + (0..7)
// halves 8..15 <- K = kb + 16*hi + 8 + (0..7)
__device__ __forceinline__ v16h load_a_frag(const _Float16* rowbase, int kb, int hi) {
  HF16x16 u;
  u.h[0] = *(const v8h*)(rowbase + kb + 16 * hi);
  u.h[1] = *(const v8h*)(rowbase + kb + 16 * hi + 8);
  return u.v;
}

// ---------------------------------------------------------------------------
// Stage 1: time LSTM.  Block = 256 threads (8 waves), 16 sequences per block.
// Wave w owns gate columns [w*64, w*64+64).  Whole Wih/Whh kept as register-
// resident B fragments; reused across all 128 time steps.
// ---------------------------------------------------------------------------
__global__ __launch_bounds__(256, 1)
void time_lstm_wmma(const float* __restrict__ x,
                    const float* __restrict__ Wih, const float* __restrict__ Whh,
                    const float* __restrict__ bih, const float* __restrict__ bhh,
                    _Float16* __restrict__ th)
{
  __shared__ _Float16 xbuf[16][KX];     // x_t tile, f16, padded K
  __shared__ _Float16 hbuf[16][HID];    // h_{t-1}, f16
  __shared__ float    gact[16][G1];     // activated gates, f32

  const int tid  = threadIdx.x;
  const int wave = tid >> 5;
  const int lane = tid & 31;
  const int lo   = lane & 15;
  const int hi   = lane >> 4;
  const int row0 = blockIdx.x * 16;

  // ---- Preload B fragments (B[k][n] = W[n][k], f32 -> f16).  Layout:
  // element j of fragment <-> K = kt*32 + 16*hi + j, N = nt*16 + lo. ----
  v16h Bx[3][4];   // x-path, K = 96 (padded from 80)
  v16h Bh[4][4];   // h-path, K = 128
  #pragma unroll
  for (int kt = 0; kt < 3; ++kt)
    #pragma unroll
    for (int nt = 0; nt < 4; ++nt) {
      const int n  = wave * 64 + nt * 16 + lo;
      const int kb = kt * 32 + hi * 16;
      #pragma unroll
      for (int j = 0; j < 16; ++j) {
        const int k = kb + j;
        Bx[kt][nt][j] = (_Float16)((k < IN_F) ? Wih[n * IN_F + k] : 0.0f);
      }
    }
  #pragma unroll
  for (int kt = 0; kt < 4; ++kt)
    #pragma unroll
    for (int nt = 0; nt < 4; ++nt) {
      const int n  = wave * 64 + nt * 16 + lo;
      const int kb = kt * 32 + hi * 16;
      #pragma unroll
      for (int j = 0; j < 16; ++j)
        Bh[kt][nt][j] = (_Float16)Whh[n * HID + kb + j];
    }

  float bias[4];
  #pragma unroll
  for (int nt = 0; nt < 4; ++nt) {
    const int n = wave * 64 + nt * 16 + lo;
    bias[nt] = bih[n] + bhh[n];
  }

  // cell state: thread owns elements e = tid*8+q -> (m = e>>7, j = e&127)
  float cst[8];
  #pragma unroll
  for (int q = 0; q < 8; ++q) cst[q] = 0.0f;

  // zero h_0 and the padded x columns
  for (int idx = tid; idx < 16 * HID; idx += 256)
    ((_Float16*)hbuf)[idx] = (_Float16)0.0f;
  { const int r = tid >> 4, c = IN_F + (tid & 15);   // 16 rows x 16 pad cols
    xbuf[r][c] = (_Float16)0.0f; }
  __syncthreads();

  for (int t = 0; t < TT; ++t) {
    // ---- stage x_t tile, f32 -> f16; prefetch next step's tile ----
    for (int idx = tid; idx < 16 * IN_F; idx += 256) {
      const int r = idx / IN_F, i = idx - r * IN_F;
      const size_t base = ((size_t)(row0 + r) * TT + t) * IN_F;
      xbuf[r][i] = (_Float16)x[base + i];
      if (i == 0 && t + 1 < TT)
        __builtin_prefetch(&x[base + IN_F], 0, 3);   // global_prefetch_b8
    }
    __syncthreads();

    // ---- load ALL 7 A fragments up front (14x ds_load_b128 in flight) ----
    v16h afr[7];
    #pragma unroll
    for (int kt = 0; kt < 3; ++kt)
      afr[kt] = load_a_frag(&xbuf[lo][0], kt * 32, hi);
    #pragma unroll
    for (int kt = 0; kt < 4; ++kt)
      afr[3 + kt] = load_a_frag(&hbuf[lo][0], kt * 32, hi);

    // ---- g = bias + x_t*Wih^T + h*Whh^T via back-to-back WMMA chain ----
    v8f acc[4];
    #pragma unroll
    for (int nt = 0; nt < 4; ++nt)
      #pragma unroll
      for (int r = 0; r < 8; ++r) acc[nt][r] = bias[nt];

    #pragma unroll
    for (int kt = 0; kt < 3; ++kt)
      #pragma unroll
      for (int nt = 0; nt < 4; ++nt)
        acc[nt] = __builtin_amdgcn_wmma_f32_16x16x32_f16(
            false, afr[kt], false, Bx[kt][nt], (short)0, acc[nt], false, false);
    #pragma unroll
    for (int kt = 0; kt < 4; ++kt)
      #pragma unroll
      for (int nt = 0; nt < 4; ++nt)
        acc[nt] = __builtin_amdgcn_wmma_f32_16x16x32_f16(
            false, afr[3 + kt], false, Bh[kt][nt], (short)0, acc[nt], false, false);

    // ---- gate activations (i,f,o: sigmoid; g: tanh) -> LDS ----
    // C layout: vgpr r, lane -> row m = 8*hi + r, col = lo (+ tile base)
    #pragma unroll
    for (int nt = 0; nt < 4; ++nt) {
      const int col = wave * 64 + nt * 16 + lo;
      const bool isg = ((col >> 7) == 2);            // cols 256..383 = g-gate
      #pragma unroll
      for (int r = 0; r < 8; ++r) {
        const float v = acc[nt][r];
        gact[hi * 8 + r][col] = isg ? tanhf(v) : sigf(v);
      }
    }
    __syncthreads();

    // ---- cell update; write h (f16) to LDS + global th ----
    #pragma unroll
    for (int q = 0; q < 8; ++q) {
      const int e = tid * 8 + q;
      const int m = e >> 7, j = e & 127;
      const float ig = gact[m][j];
      const float fg = gact[m][HID + j];
      const float gg = gact[m][2 * HID + j];
      const float og = gact[m][3 * HID + j];
      cst[q] = fg * cst[q] + ig * gg;
      const float hval = og * tanhf(cst[q]);
      const _Float16 h16 = (_Float16)hval;
      hbuf[m][j] = h16;
      th[((size_t)(row0 + m) * TT + t) * HID + j] = h16;
    }
    __syncthreads();
  }
}

// ---------------------------------------------------------------------------
// Stage 2: note LSTM (NH=2).  One wave handles 16 of the 4096 (b,t) sequences.
// Input xn[s][note][k] = th[(b*78+note)][t][k], s = b*T + t.  Gate dim 8
// padded to an N=16 WMMA tile; h*Whh^T (16x2 @ 2x8) done scalar via LDS.
// ---------------------------------------------------------------------------
__global__ __launch_bounds__(32, 1)
void note_lstm_wmma(const _Float16* __restrict__ th,
                    const float* __restrict__ Wih, const float* __restrict__ Whh,
                    const float* __restrict__ bih, const float* __restrict__ bhh,
                    float* __restrict__ out)
{
  __shared__ float hb[16][2];   // h_{n-1}, f32
  __shared__ float gb[16][8];   // activated gates

  const int lane = threadIdx.x & 31;
  const int lo = lane & 15, hi = lane >> 4;
  const int s0 = blockIdx.x * 16;

  // B fragments: Wih_n^T, [K=128][N=16] with cols 8..15 zero
  v16h Bn[4];
  #pragma unroll
  for (int kt = 0; kt < 4; ++kt) {
    const int kb = kt * 32 + hi * 16;
    #pragma unroll
    for (int j = 0; j < 16; ++j)
      Bn[kt][j] = (_Float16)((lo < 8) ? Wih[lo * HID + kb + j] : 0.0f);
  }
  const float bb  = (lo < 8) ? (bih[lo] + bhh[lo]) : 0.0f;
  const float wh0 = (lo < 8) ? Whh[lo * NHID + 0] : 0.0f;
  const float wh1 = (lo < 8) ? Whh[lo * NHID + 1] : 0.0f;

  // A-row mapping for this lane: sequence s0+lo -> (b, t)
  const int sA = s0 + lo;
  const int bA = sA >> 7;            // s / T
  const int tA = sA & (TT - 1);      // s % T

  // cell state: lane owns (m = lane>>1, j = lane&1)
  const int cm = lane >> 1, cj = lane & 1;
  float c = 0.0f;
  hb[cm][cj] = 0.0f;
  __syncthreads();

  for (int note = 0; note < NOTES; ++note) {
    const _Float16* arow = th + ((size_t)(bA * NOTES + note) * TT + tA) * HID;

    // hoist all 4 A-fragment loads (8x global b128) ahead of the WMMA chain
    v16h afr[4];
    #pragma unroll
    for (int kt = 0; kt < 4; ++kt)
      afr[kt] = load_a_frag(arow, kt * 32, hi);

    v8f acc;
    #pragma unroll
    for (int r = 0; r < 8; ++r) acc[r] = bb;

    #pragma unroll
    for (int kt = 0; kt < 4; ++kt)
      acc = __builtin_amdgcn_wmma_f32_16x16x32_f16(
          false, afr[kt], false, Bn[kt], (short)0, acc, false, false);

    // add recurrent term + activations; gates: [i0 i1 f0 f1 g0 g1 o0 o1]
    if (lo < 8) {
      #pragma unroll
      for (int r = 0; r < 8; ++r) {
        const int m = hi * 8 + r;
        const float v = acc[r] + hb[m][0] * wh0 + hb[m][1] * wh1;
        gb[m][lo] = (lo == 4 || lo == 5) ? tanhf(v) : sigf(v);
      }
    }
    __syncthreads();

    const float ig = gb[cm][cj];
    const float fg = gb[cm][2 + cj];
    const float gg = gb[cm][4 + cj];
    const float og = gb[cm][6 + cj];
    c = fg * c + ig * gg;
    const float hval = og * tanhf(c);
    __syncthreads();                 // gb/hb reads done before overwrite
    hb[cm][cj] = hval;
    out[(size_t)(s0 + cm) * (NOTES * NHID) + note * NHID + cj] = hval;
    __syncthreads();
  }
}

// ---------------------------------------------------------------------------
extern "C" void kernel_launch(void* const* d_in, const int* in_sizes, int n_in,
                              void* d_out, int out_size, void* d_ws, size_t ws_size,
                              hipStream_t stream) {
  const float* x     = (const float*)d_in[0];
  const float* Wih_t = (const float*)d_in[1];
  const float* Whh_t = (const float*)d_in[2];
  const float* bih_t = (const float*)d_in[3];
  const float* bhh_t = (const float*)d_in[4];
  const float* Wih_n = (const float*)d_in[5];
  const float* Whh_n = (const float*)d_in[6];
  const float* bih_n = (const float*)d_in[7];
  const float* bhh_n = (const float*)d_in[8];
  float* out = (float*)d_out;

  _Float16* th = (_Float16*)d_ws;    // needs 2496*128*128*2 B ~= 78 MB

  hipLaunchKernelGGL(time_lstm_wmma, dim3(NSEQ1 / 16), dim3(256), 0, stream,
                     x, Wih_t, Whh_t, bih_t, bhh_t, th);
  hipLaunchKernelGGL(note_lstm_wmma, dim3(NSEQ2 / 16), dim3(32), 0, stream,
                     th, Wih_n, Whh_n, bih_n, bhh_n, out);
}
